// FeaturePropagation_54795192762554
// MI455X (gfx1250) — matrix operand order repaired and checked by reading, not compiled, not run on hardware
//
#include <hip/hip_runtime.h>
#include <hip/hip_bf16.h>

typedef __attribute__((ext_vector_type(16))) _Float16 v16h;
typedef __attribute__((ext_vector_type(8)))  _Float16 v8h;
typedef __attribute__((ext_vector_type(8)))  float    v8f;
typedef __attribute__((ext_vector_type(4))) unsigned int u32x4;
typedef __attribute__((ext_vector_type(8)))  int i32x8;
typedef __attribute__((ext_vector_type(4)))  int i32x4;

#define B_      16
#define N1_     4096
#define N2_     1024
#define C1_     128
#define C2_     256
#define CIN0_   384   // C1 + C2
#define COUT0_  256
#define COUT1_  128
#define EPS_DIST 1e-10f
#define BN_EPS   1e-5f

#if defined(__has_builtin)
#if __has_builtin(__builtin_amdgcn_tensor_load_to_lds)
#define HAVE_TDM 1
#endif
#if __has_builtin(__builtin_amdgcn_s_wait_tensorcnt)
#define HAVE_TCNT 1
#endif
#endif

// Issue a 2-D TDM tile load (global -> LDS), descriptor per CDNA5 ISA §8.3/8.4.
// elem_code: 0=1B,1=2B,2=4B,3=8B. Dims/strides in element units. Wave-level op.
__device__ __forceinline__ void tdm_load_2d(unsigned lds_off, const void* gptr,
                                            unsigned elem_code,
                                            unsigned tensor_d0, unsigned tensor_d1,
                                            unsigned tile_d0, unsigned tile_d1,
                                            unsigned long long stride0) {
#ifdef HAVE_TDM
  unsigned long long ga = (unsigned long long)gptr;
  u32x4 g0;
  g0[0] = 1u;                                          // count=1 (valid user desc)
  g0[1] = lds_off;                                     // lds_addr (bytes)
  g0[2] = (unsigned)ga;                                // global_addr[31:0]
  g0[3] = (unsigned)((ga >> 32) & 0x01FFFFFFu) | (2u << 30);  // addr[56:32] | type=2
  i32x8 g1;
  g1[0] = (int)(elem_code << 16);                      // wg_mask=0, data_size
  g1[1] = (int)((tensor_d0 & 0xFFFFu) << 16);          // tensor_dim0[15:0]
  g1[2] = (int)((tensor_d0 >> 16) | ((tensor_d1 & 0xFFFFu) << 16));
  g1[3] = (int)((tensor_d1 >> 16) | (tile_d0 << 16));  // tile_dim0
  g1[4] = (int)(tile_d1 & 0xFFFFu);                    // tile_dim1 (tile_dim2=0)
  g1[5] = (int)(unsigned)stride0;                      // tensor_dim0_stride[31:0]
  g1[6] = (int)((stride0 >> 32) & 0xFFFFu);            // stride[47:32], dim1_stride=0
  g1[7] = 0;
  i32x4 z4 = {0, 0, 0, 0};
#if defined(__clang_major__) && __clang_major__ >= 23
  i32x8 z8 = {0, 0, 0, 0, 0, 0, 0, 0};
  __builtin_amdgcn_tensor_load_to_lds(g0, g1, z4, z4, z8, 0);
#else
  __builtin_amdgcn_tensor_load_to_lds(g0, g1, z4, z4, 0);
#endif
#endif
}

__device__ __forceinline__ void tdm_wait() {
#ifdef HAVE_TCNT
  __builtin_amdgcn_s_wait_tensorcnt(0);
#endif
}

// ---------------------------------------------------------------- weights -> f16
__global__ void k_prep_weights(const float* __restrict__ W0, const float* __restrict__ W1,
                               _Float16* __restrict__ W0h, _Float16* __restrict__ W1h) {
  int i = blockIdx.x * blockDim.x + threadIdx.x;
  if (i < COUT0_ * CIN0_)  W0h[i] = (_Float16)W0[i];
  if (i < COUT1_ * COUT0_) W1h[i] = (_Float16)W1[i];
}

// ---------------------------------------------------------------- 3-NN + weights
__global__ __launch_bounds__(128)
void k_knn(const float* __restrict__ xyz1, const float* __restrict__ xyz2,
           int* __restrict__ idx, float* __restrict__ wgt) {
  __shared__ float s[N2_ * 3];
  int b = blockIdx.y;
  int n = blockIdx.x * blockDim.x + threadIdx.x;
  for (int i = threadIdx.x; i < N2_ * 3; i += blockDim.x)
    s[i] = xyz2[(size_t)b * N2_ * 3 + i];
  __syncthreads();

  float px = xyz1[((size_t)b * N1_ + n) * 3 + 0];
  float py = xyz1[((size_t)b * N1_ + n) * 3 + 1];
  float pz = xyz1[((size_t)b * N1_ + n) * 3 + 2];

  float d0 = 3.4e38f, d1 = 3.4e38f, d2 = 3.4e38f;
  int   i0 = 0, i1 = 0, i2 = 0;
  for (int j = 0; j < N2_; ++j) {
    float dx = px - s[j * 3 + 0];
    float dy = py - s[j * 3 + 1];
    float dz = pz - s[j * 3 + 2];
    float d  = fmaxf(dx * dx + dy * dy + dz * dz, EPS_DIST);
    if (d < d0)      { d2 = d1; i2 = i1; d1 = d0; i1 = i0; d0 = d; i0 = j; }
    else if (d < d1) { d2 = d1; i2 = i1; d1 = d;  i1 = j; }
    else if (d < d2) { d2 = d;  i2 = j; }
  }
  float w0 = 1.f / d0, w1 = 1.f / d1, w2 = 1.f / d2;
  float inv = 1.f / (w0 + w1 + w2);
  size_t base = ((size_t)b * N1_ + n) * 3;
  idx[base + 0] = i0; idx[base + 1] = i1; idx[base + 2] = i2;
  wgt[base + 0] = w0 * inv; wgt[base + 1] = w1 * inv; wgt[base + 2] = w2 * inv;
}

// ---------------------------------------------------------------- interpolation gather
__global__ __launch_bounds__(256)
void k_interp(const float* __restrict__ P2, const int* __restrict__ idx,
              const float* __restrict__ wgt, float* __restrict__ interp) {
  __shared__ int   si[64 * 3];
  __shared__ float sw[64 * 3];
  int b  = blockIdx.y;
  int n0 = blockIdx.x * 64;
  if (threadIdx.x < 192) {
    size_t base = ((size_t)b * N1_ + n0) * 3 + threadIdx.x;
    si[threadIdx.x] = idx[base];
    sw[threadIdx.x] = wgt[base];
  }
  __syncthreads();
  int nl = threadIdx.x & 63;
  int cb = threadIdx.x >> 6;
  int   i0 = si[nl * 3 + 0], i1 = si[nl * 3 + 1], i2 = si[nl * 3 + 2];
  float w0 = sw[nl * 3 + 0], w1 = sw[nl * 3 + 1], w2 = sw[nl * 3 + 2];
  for (int c = cb; c < C2_; c += 4) {
    const float* p = P2 + ((size_t)b * C2_ + c) * N2_;
    interp[((size_t)b * C2_ + c) * N1_ + n0 + nl] = w0 * p[i0] + w1 * p[i1] + w2 * p[i2];
  }
}

// ================================================================ GEMM core
// Block: 128 thr (4 waves). Tile: M=64 (waves stacked), N=64, K stepped by 32.
// W tile staged in LDS via TDM; activation slab staged transposed [n][k] (stride 36)
// so B fragments are 4x ds_load_b128 per 16x16 tile, conflict-free.

#define XT_STRIDE 36

__device__ __forceinline__ void gemm_store_tile(float (*sXT)[XT_STRIDE], int tid,
                                                const float4* r) {
#pragma unroll
  for (int i = 0; i < 4; ++i) {
    int c = tid + i * 128;        // chunk id, 512 chunks of 4 floats
    int k = c >> 4;               // 0..31
    int n = (c & 15) * 4;         // 0..60
    sXT[n + 0][k] = r[i].x;
    sXT[n + 1][k] = r[i].y;
    sXT[n + 2][k] = r[i].z;
    sXT[n + 3][k] = r[i].w;
  }
}

__device__ __forceinline__ void gemm_load_tile(const float* __restrict__ src, int tid,
                                               float4* r) {
#pragma unroll
  for (int i = 0; i < 4; ++i) {
    int c = tid + i * 128;
    int k = c >> 4;
    int n = (c & 15) * 4;
    r[i] = *(const float4*)(src + (size_t)k * N1_ + n);
  }
}

template<int CIN>
__device__ __forceinline__ void gemm_compute(const _Float16* sW, float (*sXT)[XT_STRIDE],
                                             int k0, int mloc, int kA, int kB, int ncol,
                                             v8f* acc) {
  const v8h* pa = (const v8h*)(sW + (size_t)mloc * CIN + k0 + 2 * kA);
  v8h alo = pa[0];      // K = 2*kA .. 2*kA+7
  v8h ahi = pa[1];      // K = 2*kA+8 .. 2*kA+15
  v16h a = __builtin_shufflevector(alo, ahi, 0, 1, 2, 3, 4, 5, 6, 7,
                                   8, 9, 10, 11, 12, 13, 14, 15);
#pragma unroll
  for (int nt = 0; nt < 4; ++nt) {
    int nn = nt * 16 + ncol;
    const float4* pq = (const float4*)(&sXT[nn][kB]);
    float4 q0 = pq[0], q1 = pq[1], q2 = pq[2], q3 = pq[3];
    v16h bb;
    bb[0]  = (_Float16)q0.x; bb[1]  = (_Float16)q0.y;
    bb[2]  = (_Float16)q0.z; bb[3]  = (_Float16)q0.w;
    bb[4]  = (_Float16)q1.x; bb[5]  = (_Float16)q1.y;
    bb[6]  = (_Float16)q1.z; bb[7]  = (_Float16)q1.w;
    bb[8]  = (_Float16)q2.x; bb[9]  = (_Float16)q2.y;
    bb[10] = (_Float16)q2.z; bb[11] = (_Float16)q2.w;
    bb[12] = (_Float16)q3.x; bb[13] = (_Float16)q3.y;
    bb[14] = (_Float16)q3.z; bb[15] = (_Float16)q3.w;
    acc[nt] = __builtin_amdgcn_wmma_f32_16x16x32_f16(
        false, a, false, bb, (short)0, acc[nt], false, false);
  }
}

// ---------------------------------------------------------------- GEMM0
__global__ __launch_bounds__(128)
void k_gemm0(const _Float16* __restrict__ W0h, const float* __restrict__ P1,
             const float* __restrict__ interp, const float* __restrict__ bias0,
             float* __restrict__ y0) {
  __shared__ __align__(16) _Float16 sW[64 * CIN0_];        // 48 KB
  __shared__ __align__(16) float sXT[2][64][XT_STRIDE];    // 18 KB

  const int tid  = threadIdx.x;
  const int b    = blockIdx.z;
  const int lane = tid & 31;
  const int wave = tid >> 5;
  const int mblk = blockIdx.y * 64;
  const int nbase = blockIdx.x * 64;
  const int kA   = (lane >> 4) * 8;
  const int kB   = (lane >> 4) * 16;
  const int ncol = lane & 15;
  const int mloc = wave * 16 + ncol;   // row within the 64-row W tile for A-frag

  // Stage W tile [mblk..mblk+63] x CIN0_ via Tensor Data Mover (wave 0 issues).
  if (tid < 32) {
#ifdef HAVE_TDM
    tdm_load_2d((unsigned)(size_t)&sW[0], W0h + (size_t)mblk * CIN0_,
                /*2B*/1, CIN0_, 64, CIN0_, 64, CIN0_);
#else
    for (int i = lane; i < 64 * CIN0_ / 8; i += 32)
      ((v8h*)sW)[i] = ((const v8h*)(W0h + (size_t)mblk * CIN0_))[i];
#endif
  }

  const float* Xlo = P1     + (size_t)b * C1_ * N1_ + nbase;
  const float* Xhi = interp + (size_t)b * C2_ * N1_ + nbase;

  const int NK = CIN0_ / 32;   // 12 (first 4 from points1, rest from interp)
  float4 rr[2][4];
  gemm_load_tile(Xlo, tid, rr[0]);

  tdm_wait();
  __syncthreads();   // sW ready

  v8f acc[4] = {v8f{}, v8f{}, v8f{}, v8f{}};

#pragma unroll 2
  for (int ks = 0; ks < NK; ++ks) {
    gemm_store_tile(sXT[ks & 1], tid, rr[ks & 1]);
    if (ks + 1 < NK) {
      const float* src = (ks + 1 < 4) ? (Xlo + (size_t)(ks + 1) * 32 * N1_)
                                      : (Xhi + (size_t)(ks + 1 - 4) * 32 * N1_);
      gemm_load_tile(src, tid, rr[(ks + 1) & 1]);
      if (ks + 2 < NK) {
        const float* pf = (ks + 2 < 4) ? (Xlo + (size_t)(ks + 2) * 32 * N1_)
                                       : (Xhi + (size_t)(ks + 2 - 4) * 32 * N1_);
        __builtin_prefetch(pf + tid * 16, 0, 1);
      }
    }
    __syncthreads();
    gemm_compute<CIN0_>(sW, sXT[ks & 1], ks * 32, mloc, kA, kB, ncol, acc);
  }

#pragma unroll
  for (int r = 0; r < 8; ++r) {
    int ch = mblk + wave * 16 + r + 8 * (lane >> 4);
    float bv = bias0[ch];
    float* dst = y0 + ((size_t)b * COUT0_ + ch) * N1_ + nbase + ncol;
    dst[0]  = acc[0][r] + bv;
    dst[16] = acc[1][r] + bv;
    dst[32] = acc[2][r] + bv;
    dst[48] = acc[3][r] + bv;
  }
}

// ---------------------------------------------------------------- BN stats -> (scale, shift)
__global__ __launch_bounds__(256)
void k_stats(const float* __restrict__ y, const float* __restrict__ g,
             const float* __restrict__ be, float* __restrict__ scale,
             float* __restrict__ shift, int C) {
  __shared__ float ss[256];
  __shared__ float sq[256];
  int c = blockIdx.x;
  float s = 0.f, q = 0.f;
  for (int b = 0; b < B_; ++b) {
    const float* p = y + ((size_t)b * C + c) * N1_;
    for (int i = threadIdx.x; i < N1_; i += 256) { float v = p[i]; s += v; q += v * v; }
  }
  ss[threadIdx.x] = s; sq[threadIdx.x] = q;
  __syncthreads();
  for (int o = 128; o > 0; o >>= 1) {
    if (threadIdx.x < (unsigned)o) {
      ss[threadIdx.x] += ss[threadIdx.x + o];
      sq[threadIdx.x] += sq[threadIdx.x + o];
    }
    __syncthreads();
  }
  if (threadIdx.x == 0) {
    float inv = 1.f / (float)(B_ * N1_);
    float m   = ss[0] * inv;
    float var = sq[0] * inv - m * m;
    float rs  = rsqrtf(var + BN_EPS);
    scale[c] = g[c] * rs;
    shift[c] = be[c] - g[c] * rs * m;
  }
}

// ---------------------------------------------------------------- GEMM1 (BN0+ReLU fused at staging)
__global__ __launch_bounds__(128)
void k_gemm1(const _Float16* __restrict__ W1h, const float* __restrict__ y0,
             const float* __restrict__ scale0, const float* __restrict__ shift0,
             const float* __restrict__ bias1, float* __restrict__ y1) {
  __shared__ __align__(16) _Float16 sW[64 * COUT0_];       // 32 KB
  __shared__ __align__(16) float sXT[2][64][XT_STRIDE];    // 18 KB

  const int tid  = threadIdx.x;
  const int b    = blockIdx.z;
  const int lane = tid & 31;
  const int wave = tid >> 5;
  const int mblk = blockIdx.y * 64;
  const int nbase = blockIdx.x * 64;
  const int kA   = (lane >> 4) * 8;
  const int kB   = (lane >> 4) * 16;
  const int ncol = lane & 15;
  const int mloc = wave * 16 + ncol;

  if (tid < 32) {
#ifdef HAVE_TDM
    tdm_load_2d((unsigned)(size_t)&sW[0], W1h + (size_t)mblk * COUT0_,
                /*2B*/1, COUT0_, 64, COUT0_, 64, COUT0_);
#else
    for (int i = lane; i < 64 * COUT0_ / 8; i += 32)
      ((v8h*)sW)[i] = ((const v8h*)(W1h + (size_t)mblk * COUT0_))[i];
#endif
  }

  const float* Y = y0 + (size_t)b * COUT0_ * N1_ + nbase;
  const int NK = COUT0_ / 32;   // 8
  float4 rr[2][4];
  gemm_load_tile(Y, tid, rr[0]);

  tdm_wait();
  __syncthreads();

  v8f acc[4] = {v8f{}, v8f{}, v8f{}, v8f{}};

#pragma unroll 2
  for (int ks = 0; ks < NK; ++ks) {
    // apply BN0 + ReLU while staging into LDS (once per element, not per reader)
    {
      float4* r = rr[ks & 1];
#pragma unroll
      for (int i = 0; i < 4; ++i) {
        int c = tid + i * 128;
        int kk = ks * 32 + (c >> 4);
        float sc = scale0[kk], sh = shift0[kk];
        r[i].x = fmaxf(sc * r[i].x + sh, 0.f);
        r[i].y = fmaxf(sc * r[i].y + sh, 0.f);
        r[i].z = fmaxf(sc * r[i].z + sh, 0.f);
        r[i].w = fmaxf(sc * r[i].w + sh, 0.f);
      }
    }
    gemm_store_tile(sXT[ks & 1], tid, rr[ks & 1]);
    if (ks + 1 < NK) {
      gemm_load_tile(Y + (size_t)(ks + 1) * 32 * N1_, tid, rr[(ks + 1) & 1]);
      if (ks + 2 < NK)
        __builtin_prefetch(Y + (size_t)(ks + 2) * 32 * N1_ + tid * 16, 0, 1);
    }
    __syncthreads();
    gemm_compute<COUT0_>(sW, sXT[ks & 1], ks * 32, mloc, kA, kB, ncol, acc);
  }

#pragma unroll
  for (int r = 0; r < 8; ++r) {
    int ch = mblk + wave * 16 + r + 8 * (lane >> 4);
    float bv = bias1[ch];
    float* dst = y1 + ((size_t)b * COUT1_ + ch) * N1_ + nbase + ncol;
    dst[0]  = acc[0][r] + bv;
    dst[16] = acc[1][r] + bv;
    dst[32] = acc[2][r] + bv;
    dst[48] = acc[3][r] + bv;
  }
}

// ---------------------------------------------------------------- final BN1 + ReLU
__global__ __launch_bounds__(256)
void k_apply(const float* __restrict__ y1, const float* __restrict__ scale,
             const float* __restrict__ shift, float* __restrict__ out) {
  size_t i = (size_t)blockIdx.x * blockDim.x + threadIdx.x;
  size_t total = (size_t)B_ * COUT1_ * N1_;
  if (i >= total) return;
  int c = (int)((i / N1_) % COUT1_);
  out[i] = fmaxf(scale[c] * y1[i] + shift[c], 0.f);
}

// ================================================================ launcher
extern "C" void kernel_launch(void* const* d_in, const int* in_sizes, int n_in,
                              void* d_out, int out_size, void* d_ws, size_t ws_size,
                              hipStream_t stream) {
  const float* xyz1    = (const float*)d_in[0];
  const float* xyz2    = (const float*)d_in[1];
  const float* points1 = (const float*)d_in[2];
  const float* points2 = (const float*)d_in[3];
  const float* W0      = (const float*)d_in[4];
  const float* b0      = (const float*)d_in[5];
  const float* g0      = (const float*)d_in[6];
  const float* be0     = (const float*)d_in[7];
  const float* W1      = (const float*)d_in[8];
  const float* b1      = (const float*)d_in[9];
  const float* g1      = (const float*)d_in[10];
  const float* be1     = (const float*)d_in[11];
  float* out = (float*)d_out;

  char* ws = (char*)d_ws;
  _Float16* W0h    = (_Float16*)(ws + 0);            // 196608 B
  _Float16* W1h    = (_Float16*)(ws + 196608);       //  65536 B
  int*      idx    = (int*)     (ws + 262144);       // 786432 B
  float*    wgt    = (float*)   (ws + 1048576);      // 786432 B
  float*    scale0 = (float*)   (ws + 1835008);
  float*    shift0 = (float*)   (ws + 1836032);
  float*    scale1 = (float*)   (ws + 1837056);
  float*    shift1 = (float*)   (ws + 1837568);
  float*    interp = (float*)   (ws + 2097152);      // 64 MB
  float*    y0     = (float*)   (ws + 2097152 + 67108864); // 64 MB
  float*    y1     = interp;                         // reuse

  k_prep_weights<<<(COUT0_ * CIN0_ + 255) / 256, 256, 0, stream>>>(W0, W1, W0h, W1h);
  k_knn   <<<dim3(N1_ / 128, B_), 128, 0, stream>>>(xyz1, xyz2, idx, wgt);
  k_interp<<<dim3(N1_ / 64,  B_), 256, 0, stream>>>(points2, idx, wgt, interp);
  k_gemm0 <<<dim3(N1_ / 64, COUT0_ / 64, B_), 128, 0, stream>>>(W0h, points1, interp, b0, y0);
  k_stats <<<COUT0_, 256, 0, stream>>>(y0, g0, be0, scale0, shift0, COUT0_);
  k_gemm1 <<<dim3(N1_ / 64, COUT1_ / 64, B_), 128, 0, stream>>>(W1h, y0, scale0, shift0, b1, y1);
  k_stats <<<COUT1_, 256, 0, stream>>>(y1, g1, be1, scale1, shift1, COUT1_);
  k_apply <<<((size_t)B_ * COUT1_ * N1_ + 255) / 256, 256, 0, stream>>>(y1, scale1, shift1, out);
}
